// PointnetSAModuleVotes_3315714753183
// MI455X (gfx1250) — compile-verified, matrix operand order checked by reading, not run on hardware
//
#include <hip/hip_runtime.h>
#include <hip/hip_bf16.h>

typedef __attribute__((ext_vector_type(16))) _Float16 v16h;
typedef __attribute__((ext_vector_type(8)))  _Float16 v8h;
typedef __attribute__((ext_vector_type(2)))  _Float16 v2h;
typedef __attribute__((ext_vector_type(8)))  float    v8f;

#define BATCH   8
#define NPTS    8192
#define NPOINT  2048
#define NSAMPLE 64
#define RADIUSF 0.4f
#define EPSV    1e-5f

// packed-weight fragment counts: L0 = 4 mtiles * 3 ksteps, L1 = 4*2, L2 = 8*2
#define PW_L0_OFF 0
#define PW_L1_OFF (12 * 512)
#define PW_L2_OFF (20 * 512)
#define PW_HALVES (36 * 512)            // 18432 halves = 36864 bytes
#define BNC_BYTE_OFF 36864              // float2 bn coeffs after packed weights

// ---------------------------------------------------------------------------
// Kernel 1: farthest point sampling. Two-level argmax: 5 shfl_xor steps within
// each wave32 (no barrier), then one wave reduces the 32 partials -> only three
// __syncthreads per iteration on the 2047-long serial chain.
// ---------------------------------------------------------------------------
__global__ __launch_bounds__(1024) void fps_kernel(const float* __restrict__ xyz,
                                                   float* __restrict__ new_xyz,
                                                   float* __restrict__ inds_out) {
  const int b    = blockIdx.x;
  const int tid  = threadIdx.x;
  const int lane = tid & 31;
  const int wv   = tid >> 5;
  const float* X = xyz + (size_t)b * NPTS * 3;

  __shared__ float dist[NPTS];
  __shared__ float wmaxv[32];
  __shared__ int   wmaxi[32];
  __shared__ float cur[3];

  const float x0 = X[0], y0 = X[1], z0 = X[2];
  for (int j = tid; j < NPTS; j += 1024) {
    float dx = X[j * 3 + 0] - x0;
    float dy = X[j * 3 + 1] - y0;
    float dz = X[j * 3 + 2] - z0;
    dist[j] = dx * dx + dy * dy + dz * dz;
  }
  if (tid == 0) {
    size_t o = (size_t)b * NPOINT;
    new_xyz[o * 3 + 0] = x0;
    new_xyz[o * 3 + 1] = y0;
    new_xyz[o * 3 + 2] = z0;
    inds_out[o] = 0.0f;
  }
  __syncthreads();

  for (int it = 1; it < NPOINT; ++it) {
    float bv = -1.0f;
    int   bi = 0x7fffffff;
    for (int j = tid; j < NPTS; j += 1024) {
      float d = dist[j];
      if (d > bv) { bv = d; bi = j; }   // ascending j -> first-max tie-break
    }
    // in-wave argmax (tie-break: lowest index)
#pragma unroll
    for (int sh = 16; sh >= 1; sh >>= 1) {
      float ov = __shfl_xor(bv, sh, 32);
      int   oi = __shfl_xor(bi, sh, 32);
      if (ov > bv || (ov == bv && oi < bi)) { bv = ov; bi = oi; }
    }
    if (lane == 0) { wmaxv[wv] = bv; wmaxi[wv] = bi; }
    __syncthreads();
    if (tid < 32) {
      bv = wmaxv[tid];
      bi = wmaxi[tid];
#pragma unroll
      for (int sh = 16; sh >= 1; sh >>= 1) {
        float ov = __shfl_xor(bv, sh, 32);
        int   oi = __shfl_xor(bi, sh, 32);
        if (ov > bv || (ov == bv && oi < bi)) { bv = ov; bi = oi; }
      }
      if (tid == 0) {
        int nxt = bi;
        float px = X[nxt * 3 + 0];
        float py = X[nxt * 3 + 1];
        float pz = X[nxt * 3 + 2];
        cur[0] = px; cur[1] = py; cur[2] = pz;
        size_t o = (size_t)b * NPOINT + it;
        new_xyz[o * 3 + 0] = px;
        new_xyz[o * 3 + 1] = py;
        new_xyz[o * 3 + 2] = pz;
        inds_out[o] = (float)nxt;
      }
    }
    __syncthreads();
    const float cx = cur[0], cy = cur[1], cz = cur[2];
    for (int j = tid; j < NPTS; j += 1024) {
      float dx = X[j * 3 + 0] - cx;
      float dy = X[j * 3 + 1] - cy;
      float dz = X[j * 3 + 2] - cz;
      float d2 = dx * dx + dy * dy + dz * dz;
      float d  = dist[j];
      dist[j] = d2 < d ? d2 : d;
    }
    __syncthreads();
  }
}

// ---------------------------------------------------------------------------
// Kernel 0: pack weights to f16 A-fragment lane order + fold BN coefficients.
// ---------------------------------------------------------------------------
__global__ __launch_bounds__(256) void pack_kernel(
    const float* __restrict__ W0, const float* __restrict__ W1,
    const float* __restrict__ W2,
    const float* __restrict__ g0, const float* __restrict__ b0,
    const float* __restrict__ m0, const float* __restrict__ v0,
    const float* __restrict__ g1, const float* __restrict__ b1,
    const float* __restrict__ m1, const float* __restrict__ v1,
    const float* __restrict__ g2, const float* __restrict__ b2,
    const float* __restrict__ m2, const float* __restrict__ v2,
    _Float16* __restrict__ pw, float2* __restrict__ bnc) {
  const int tid = threadIdx.x;

  for (int e = tid; e < PW_HALVES; e += 256) {
    int f    = e >> 9;      // fragment index 0..35
    int r    = e & 511;
    int lane = r >> 4;
    int i    = r & 15;
    const float* W;
    int ldw, kreal, nk, fl = f;
    if (fl < 12)      { W = W0; ldw = 67; kreal = 67; nk = 3; }
    else if (fl < 20) { fl -= 12; W = W1; ldw = 64; kreal = 64; nk = 2; }
    else              { fl -= 20; W = W2; ldw = 64; kreal = 64; nk = 2; }
    int mtile = fl / nk;
    int ks    = fl - mtile * nk;
    int m  = mtile * 16 + (lane & 15);
    int kb = (lane < 16) ? 0 : 8;
    int kl = (i < 8 ? i : i + 8) + kb;
    int k  = ks * 32 + kl;
    float w = (k < kreal) ? W[m * ldw + k] : 0.0f;
    pw[e] = (_Float16)w;
  }

  // folded BN: y = x*a + c, a = g*rsqrt(v+eps), c = b - mean*a (pre-ReLU)
  {
    int ch = tid;            // 0..255 (64 + 64 + 128)
    const float *g, *be, *mu, *va;
    int m;
    if (ch < 64)       { g = g0; be = b0; mu = m0; va = v0; m = ch; }
    else if (ch < 128) { g = g1; be = b1; mu = m1; va = v1; m = ch - 64; }
    else               { g = g2; be = b2; mu = m2; va = v2; m = ch - 128; }
    float a = g[m] * rsqrtf(va[m] + EPSV);
    bnc[ch] = make_float2(a, be[m] - mu[m] * a);
  }
}

// ---------------------------------------------------------------------------
// Fragment helpers.
// ---------------------------------------------------------------------------
__device__ inline v16h ld_frag16(const _Float16* __restrict__ p) {
  v8h lo = *(const v8h*)p;
  v8h hi = *(const v8h*)(p + 8);
  return __builtin_shufflevector(lo, hi, 0, 1, 2, 3, 4, 5, 6, 7,
                                 8, 9, 10, 11, 12, 13, 14, 15);
}

// position of element (k,n) inside a swizzled B buffer (frag = (k>>5)*4 + n>>4)
__device__ inline int bpos(int k, int n) {
  int f    = ((k >> 5) << 2) + (n >> 4);
  int lane = (n & 15) + (((k >> 3) & 1) << 4);
  int i    = (k & 7) + (((k >> 4) & 1) << 3);
  return ((f << 5) + lane) * 16 + i;
}

template <int NK>
__device__ inline void mlp_tile(const _Float16* __restrict__ Xs,
                                const _Float16* __restrict__ PA,  // frag base for this mtile
                                int lane, v8f acc[4]) {
  v8f z = {};
#pragma unroll
  for (int t = 0; t < 4; ++t) acc[t] = z;
#pragma unroll
  for (int ks = 0; ks < NK; ++ks) {
    v16h a = ld_frag16(PA + ks * 512 + lane * 16);
#pragma unroll
    for (int t = 0; t < 4; ++t) {
      v16h bf = ld_frag16(Xs + (((ks * 4 + t) * 32) + lane) * 16);
      acc[t] = __builtin_amdgcn_wmma_f32_16x16x32_f16(
          false, a, false, bf, (short)0, acc[t], false, false);
    }
  }
}

__device__ inline void bn_relu_store_sw(const v8f acc[4],
                                        const float2* __restrict__ bnl, int chbase,
                                        int mbase, int lane,
                                        _Float16* __restrict__ Xs) {
  const int nlo = lane & 15;
  const int hi  = lane >> 4;
#pragma unroll
  for (int r = 0; r < 8; r += 2) {
    int m = mbase + hi * 8 + r;          // even
    float2 p0 = bnl[chbase + m];
    float2 p1 = bnl[chbase + m + 1];
#pragma unroll
    for (int t = 0; t < 4; ++t) {
      float u0 = fmaxf(acc[t][r] * p0.x + p0.y, 0.0f);
      float u1 = fmaxf(acc[t][r + 1] * p1.x + p1.y, 0.0f);
      v2h pr;
      pr[0] = (_Float16)u0;
      pr[1] = (_Float16)u1;
      *(v2h*)&Xs[bpos(m, t * 16 + nlo)] = pr;   // paired rows -> adjacent halves
    }
  }
}

__device__ inline float gather_val(int c, int j, const float* __restrict__ Xb,
                                   const float* __restrict__ Fb,
                                   float cx, float cy, float cz) {
  if (c == 0) return Xb[j * 3 + 0] - cx;
  if (c == 1) return Xb[j * 3 + 1] - cy;
  if (c == 2) return Xb[j * 3 + 2] - cz;
  if (c < 67) return Fb[(size_t)(c - 3) * NPTS + j];
  return 0.0f;
}

// ---------------------------------------------------------------------------
// Kernel 2: fused ball-query + gather + 3-layer WMMA MLP + BN/ReLU + max.
// One 128-thread (4-wave) block per center. Activations in swizzled LDS.
// BN coefficients staged into LDS with global_load_async_to_lds_b128.
// ---------------------------------------------------------------------------
__global__ __launch_bounds__(128) void sa_mlp_kernel(
    const float* __restrict__ xyz, const float* __restrict__ feat,
    const _Float16* __restrict__ pw, const float2* __restrict__ bnc,
    const float* __restrict__ new_xyz, float* __restrict__ out_feat) {
  const int gc   = blockIdx.x;            // b * NPOINT + s
  const int b    = gc / NPOINT;
  const int s    = gc - b * NPOINT;
  const int tid  = threadIdx.x;
  const int lane = tid & 31;
  const int wid  = tid >> 5;

  __shared__ int sidx[NSAMPLE];
  __shared__ int scnt[129];
  __shared__ __align__(16) float2   bnl[256];       // async-staged BN coeffs (2KB)
  __shared__ __align__(16) _Float16 Xs0[12 * 512];  // layer0 in (96x64) / layer2 in
  __shared__ __align__(16) _Float16 Xs1[8 * 512];   // layer1 in (64x64)

  // ---- async-stage BN coefficient table into LDS (16B per lane, ASYNCcnt) ----
  {
    unsigned ldsoff = (unsigned)(uintptr_t)(&bnl[0]) + (unsigned)tid * 16u;
    unsigned goff   = (unsigned)tid * 16u;
    asm volatile("global_load_async_to_lds_b128 %0, %1, %2"
                 :: "v"(ldsoff), "v"(goff),
                    "s"((unsigned long long)(uintptr_t)bnc)
                 : "memory");
  }
  if (tid == 0) {
    __builtin_prefetch(pw, 0, 3);
    __builtin_prefetch(pw + 8192, 0, 3);
    __builtin_prefetch(pw + 16384, 0, 3);
  }

  const float cx = new_xyz[(size_t)gc * 3 + 0];
  const float cy = new_xyz[(size_t)gc * 3 + 1];
  const float cz = new_xyz[(size_t)gc * 3 + 2];
  const float* Xb = xyz  + (size_t)b * NPTS * 3;
  const float* Fb = feat + (size_t)b * 64 * NPTS;

  // ---- ball query pass 1: 64-bit hit mask per thread (one distance pass) ----
  const float r2   = RADIUSF * RADIUSF;
  const int   base = tid * (NPTS / 128);
  unsigned long long mask = 0ull;
  for (int i = 0; i < NPTS / 128; ++i) {
    int j = base + i;
    float dx = Xb[j * 3 + 0] - cx;
    float dy = Xb[j * 3 + 1] - cy;
    float dz = Xb[j * 3 + 2] - cz;
    if (dx * dx + dy * dy + dz * dz < r2) mask |= (1ull << i);
  }
  scnt[tid] = __popcll(mask);
  __syncthreads();
  if (tid == 0) {
    int run = 0;
    for (int t = 0; t < 128; ++t) { int c = scnt[t]; scnt[t] = run; run += c; }
    scnt[128] = run;
  }
  __syncthreads();
  // ---- pass 2: walk set bits (no distance recompute) ----
  {
    int off = scnt[tid];
    unsigned long long mm = mask;
    while (mm && off < NSAMPLE) {
      int i = __ffsll((long long)mm) - 1;
      mm &= mm - 1;
      sidx[off++] = base + i;
    }
  }
  __syncthreads();
  if (tid == 0) {
    int total = scnt[128];
    int lim   = total < NSAMPLE ? total : NSAMPLE;
    int fill  = (total > 0) ? sidx[0] : 0;
    for (int p = lim; p < NSAMPLE; ++p) sidx[p] = fill;
  }
  __syncthreads();

  // ---- gather: write (k,k+1) pairs straight into the swizzled B layout ----
  for (int e = tid; e < 48 * NSAMPLE; e += 128) {
    int kp = e >> 6;           // pair-row 0..47 -> k = 2*kp (even)
    int n  = e & 63;
    int k  = kp * 2;
    int j  = sidx[n];
    float u0 = gather_val(k, j, Xb, Fb, cx, cy, cz);
    float u1 = gather_val(k + 1, j, Xb, Fb, cx, cy, cz);
    v2h pr;
    pr[0] = (_Float16)u0;
    pr[1] = (_Float16)u1;
    *(v2h*)&Xs0[bpos(k, n)] = pr;
  }
  // own async BN staging must be complete before the block-wide barrier;
  // after the barrier every wave sees the full table.
  asm volatile("s_wait_asynccnt 0" ::: "memory");
  __syncthreads();

  v8f acc[4];

  // ---- layer 0: out 64, K=67 (padded 96, 3 k-steps) ----
  mlp_tile<3>(Xs0, pw + PW_L0_OFF + wid * 3 * 512, lane, acc);
  bn_relu_store_sw(acc, bnl, 0, wid * 16, lane, Xs1);
  __syncthreads();

  // ---- layer 1: out 64, K=64 (2 k-steps); writes into Xs0 frags 0..7 ----
  mlp_tile<2>(Xs1, pw + PW_L1_OFF + wid * 2 * 512, lane, acc);
  bn_relu_store_sw(acc, bnl, 64, wid * 16, lane, Xs0);
  __syncthreads();

  // ---- layer 2: out 128, K=64; BN/ReLU + max over 64 samples ----
#pragma unroll
  for (int mt = 0; mt < 2; ++mt) {
    const int mtile = wid * 2 + mt;
    const int mbase = mtile * 16;
    mlp_tile<2>(Xs0, pw + PW_L2_OFF + mtile * 2 * 512, lane, acc);
    const int hi = lane >> 4;
    float mx[8];
#pragma unroll
    for (int r = 0; r < 8; ++r) {
      int m = mbase + r + hi * 8;
      float2 p = bnl[128 + m];
      float best = -1e30f;
#pragma unroll
      for (int t = 0; t < 4; ++t) {
        float val = fmaxf(acc[t][r] * p.x + p.y, 0.0f);
        best = fmaxf(best, val);
      }
#pragma unroll
      for (int sh = 8; sh >= 1; sh >>= 1)
        best = fmaxf(best, __shfl_xor(best, sh, 32));
      mx[r] = best;
    }
    if ((lane & 15) == 0) {
#pragma unroll
      for (int r = 0; r < 8; ++r) {
        int m = mbase + r + hi * 8;
        out_feat[((size_t)b * 128 + m) * NPOINT + s] = mx[r];
      }
    }
  }
}

// ---------------------------------------------------------------------------
extern "C" void kernel_launch(void* const* d_in, const int* in_sizes, int n_in,
                              void* d_out, int out_size, void* d_ws, size_t ws_size,
                              hipStream_t stream) {
  const float* xyz  = (const float*)d_in[0];
  const float* feat = (const float*)d_in[1];
  const float* W0 = (const float*)d_in[2];
  const float* g0 = (const float*)d_in[3];
  const float* b0 = (const float*)d_in[4];
  const float* m0 = (const float*)d_in[5];
  const float* v0 = (const float*)d_in[6];
  const float* W1 = (const float*)d_in[7];
  const float* g1 = (const float*)d_in[8];
  const float* b1 = (const float*)d_in[9];
  const float* m1 = (const float*)d_in[10];
  const float* v1 = (const float*)d_in[11];
  const float* W2 = (const float*)d_in[12];
  const float* g2 = (const float*)d_in[13];
  const float* b2 = (const float*)d_in[14];
  const float* m2 = (const float*)d_in[15];
  const float* v2 = (const float*)d_in[16];

  float* out      = (float*)d_out;
  float* new_xyz  = out;                                      // B*NPOINT*3
  float* new_feat = out + (size_t)BATCH * NPOINT * 3;         // B*128*NPOINT
  float* inds_out = new_feat + (size_t)BATCH * 128 * NPOINT;  // B*NPOINT (float)

  _Float16* pw  = (_Float16*)d_ws;
  float2*   bnc = (float2*)((char*)d_ws + BNC_BYTE_OFF);

  pack_kernel<<<1, 256, 0, stream>>>(W0, W1, W2, g0, b0, m0, v0,
                                     g1, b1, m1, v1, g2, b2, m2, v2, pw, bnc);

  fps_kernel<<<BATCH, 1024, 0, stream>>>(xyz, new_xyz, inds_out);

  sa_mlp_kernel<<<BATCH * NPOINT, 128, 0, stream>>>(xyz, feat, pw, bnc,
                                                    new_xyz, new_feat);
}
